// MultiheadAttention_1554778161657
// MI455X (gfx1250) — compile-verified, matrix operand order checked by reading, not run on hardware
//
#include <hip/hip_runtime.h>
#include <hip/hip_bf16.h>

#define EMB    2048
#define HEADS  16
#define DHEAD  128
#define SLEN   2048
#define BATCH  2
#define TOKENS (BATCH * SLEN)

typedef __attribute__((ext_vector_type(16))) _Float16 v16h;
typedef __attribute__((ext_vector_type(8)))  float    v8f;
typedef __attribute__((ext_vector_type(8)))  int      v8i;
typedef __attribute__((ext_vector_type(4)))  int      i4;
typedef __attribute__((ext_vector_type(2)))  int      i2;

union FragH { i4 q[2]; v16h h; };
union FragB { i2 q[4]; v8i  b; };
union AccF  { v8f v; float f[8]; };
union AccI  { v8i v; int   f[8]; };

static __device__ __forceinline__ void wait_ds0() {
#if __has_builtin(__builtin_amdgcn_s_wait_dscnt)
  __builtin_amdgcn_s_wait_dscnt(0);
#else
  asm volatile("s_wait_dscnt 0" ::: "memory");
#endif
}

template <int N>
static __device__ __forceinline__ void wait_async() {
#if __has_builtin(__builtin_amdgcn_s_wait_asynccnt)
  __builtin_amdgcn_s_wait_asynccnt(N);
#else
  asm volatile("s_wait_asynccnt %0" ::"i"(N) : "memory");
#endif
}

static __device__ __forceinline__ void cp16_g2lds(void* lds, const void* g) {
#if __has_builtin(__builtin_amdgcn_global_load_async_to_lds_b128)
  __builtin_amdgcn_global_load_async_to_lds_b128(
      (__attribute__((address_space(1))) i4*)(g),
      (__attribute__((address_space(3))) i4*)(lds), 0, 0);
#else
  *(i4*)lds = *(const i4*)g;
#endif
}

__global__ __launch_bounds__(256) void wabs_kernel(
    const float* __restrict__ w0, const float* __restrict__ w1,
    const float* __restrict__ w2, const float* __restrict__ w3,
    float* __restrict__ partials  ) {
  const float* w = (blockIdx.y == 0) ? w0 : (blockIdx.y == 1) ? w1
                 : (blockIdx.y == 2) ? w2 : w3;
  __shared__ float red[256];
  float s = 0.f;
  for (size_t i = (size_t)blockIdx.x * 256 + threadIdx.x;
       i < (size_t)EMB * EMB; i += (size_t)256 * 256)
    s += fabsf(w[i]);
  red[threadIdx.x] = s;
  __syncthreads();
  for (int t = 128; t > 0; t >>= 1) {
    if ((int)threadIdx.x < t) red[threadIdx.x] += red[threadIdx.x + t];
    __syncthreads();
  }
  if (threadIdx.x == 0) partials[blockIdx.y * 256 + blockIdx.x] = red[0];
}

__global__ __launch_bounds__(256) void wquant_kernel(
    const float* __restrict__ w0, const float* __restrict__ w1,
    const float* __restrict__ w2, const float* __restrict__ w3,
    const float* __restrict__ partials, float* __restrict__ wsum,
    signed char* __restrict__ wqbase) {
  const int widx = blockIdx.y;
  const float* w = (widx == 0) ? w0 : (widx == 1) ? w1 : (widx == 2) ? w2 : w3;
  signed char* q = wqbase + (size_t)widx * EMB * EMB;
  float total = 0.f;
  for (int i = 0; i < 256; ++i) total += partials[widx * 256 + i];
  if (blockIdx.x == 0 && threadIdx.x == 0) wsum[widx] = total;
  const float mean = total * (1.f / ((float)EMB * (float)EMB));
  const float sw = 1.f / fmaxf(mean, 1e-5f);
  const size_t stride = (size_t)gridDim.x * 256;
  for (size_t i = (size_t)blockIdx.x * 256 + threadIdx.x;
       i < (size_t)EMB * EMB; i += stride) {
    float r = rintf(w[i] * sw);
    r = fminf(fmaxf(r, -1.f), 1.f);
    q[i] = (signed char)r;
  }
}

__global__ __launch_bounds__(256) void act_quant_kernel(
    const float* __restrict__ X, signed char* __restrict__ Xq,
    float* __restrict__ Ax) {
  __shared__ float red[256];
  const int t = blockIdx.x;
  const int tid = threadIdx.x;
  const float* row = X + (size_t)t * EMB;
  float mx = 0.f;
  for (int i = tid; i < EMB; i += 256) mx = fmaxf(mx, fabsf(row[i]));
  red[tid] = mx;
  __syncthreads();
  for (int s = 128; s > 0; s >>= 1) {
    if (tid < s) red[tid] = fmaxf(red[tid], red[tid + s]);
    __syncthreads();
  }
  const float amax = fmaxf(red[0], 1e-5f);
  const float sx = 127.f / amax;
  if (tid == 0) Ax[t] = amax * (1.f / 127.f);
  signed char* qrow = Xq + (size_t)t * EMB;
  for (int i = tid; i < EMB; i += 256) {
    float q = rintf(row[i] * sx);
    q = fminf(fmaxf(q, -128.f), 127.f);
    qrow[i] = (signed char)q;
  }
}

__global__ __launch_bounds__(256) void gemm_i8_kernel(
    const signed char* __restrict__ Xq, const float* __restrict__ Ax,
    const signed char* __restrict__ Wq, const float* __restrict__ wsum,
    int widx, int mode, void* __restrict__ outp) {
  __shared__ __align__(16) signed char Bs[2][128 * 64];
  const int tid = threadIdx.x;
  const int lane = tid & 31;
  const int wave = tid >> 5;
  const int hl = lane >> 4;
  const int ln = lane & 15;
  const int mbase = blockIdx.y * 128 + wave * 16;
  const int nbase = blockIdx.x * 128;

  const signed char* arow = Xq + (size_t)(mbase + ln) * EMB + hl * 8;

  const int c0 = tid, c1 = tid + 256;
  const int pr0 = c0 >> 2, po0 = (c0 & 3) * 16;
  const int pr1 = c1 >> 2, po1 = (c1 & 3) * 16;
  const signed char* wrow0 = Wq + (size_t)(nbase + pr0) * EMB + po0;
  const signed char* wrow1 = Wq + (size_t)(nbase + pr1) * EMB + po1;

  AccI acc[8] = {};
  cp16_g2lds(&Bs[0][pr0 * 64 + po0], wrow0);
  cp16_g2lds(&Bs[0][pr1 * 64 + po1], wrow1);
  int cur = 0;

  for (int kk = 0; kk < EMB; kk += 64) {
    if (kk + 64 < EMB) {
      cp16_g2lds(&Bs[cur ^ 1][pr0 * 64 + po0], wrow0 + kk + 64);
      cp16_g2lds(&Bs[cur ^ 1][pr1 * 64 + po1], wrow1 + kk + 64);
      wait_async<2>();
      __builtin_prefetch(arow + kk + 64, 0, 1);
    } else {
      wait_async<0>();
    }
    __syncthreads();

    FragB a;
    const signed char* ap = arow + kk;
    a.q[0] = *(const i2*)(ap);      a.q[1] = *(const i2*)(ap + 16);
    a.q[2] = *(const i2*)(ap + 32); a.q[3] = *(const i2*)(ap + 48);

    FragB b[8];
#pragma unroll
    for (int nt = 0; nt < 8; ++nt) {
      const signed char* bp = &Bs[cur][(nt * 16 + ln) * 64 + hl * 8];
      b[nt].q[0] = *(const i2*)(bp);      b[nt].q[1] = *(const i2*)(bp + 16);
      b[nt].q[2] = *(const i2*)(bp + 32); b[nt].q[3] = *(const i2*)(bp + 48);
    }
#pragma unroll
    for (int nt = 0; nt < 8; ++nt)
      acc[nt].v = __builtin_amdgcn_wmma_i32_16x16x64_iu8(
          true, a.b, true, b[nt].b, acc[nt].v, false, false);

    __syncthreads();
    cur ^= 1;
  }

  const float wsc =
      fmaxf(wsum[widx] * (1.f / ((float)EMB * (float)EMB)), 1e-5f);
#pragma unroll
  for (int r = 0; r < 8; ++r) {
    const int row = mbase + r + hl * 8;
    float s = Ax[row] * wsc;
    if (mode == 0) s *= 0.08838834764831845f;
    const int bb = row / SLEN, ss = row % SLEN;
#pragma unroll
    for (int nt = 0; nt < 8; ++nt) {
      const int col = nbase + nt * 16 + ln;
      const float y = (float)acc[nt].f[r] * s;
      if (mode <= 1) {
        const int h = col / DHEAD, d = col % DHEAD;
        ((_Float16*)outp)[(((size_t)(bb * HEADS + h)) * SLEN + ss) * DHEAD + d] =
            (_Float16)y;
      } else if (mode == 2) {
        const int h = col / DHEAD, d = col % DHEAD;
        ((_Float16*)outp)[(((size_t)(bb * HEADS + h)) * DHEAD + d) * SLEN + ss] =
            (_Float16)y;
      } else {
        ((float*)outp)[(size_t)row * EMB + col] = y;
      }
    }
  }
}

__global__ __launch_bounds__(256) void attn_kernel(
    const _Float16* __restrict__ Q, const _Float16* __restrict__ K,
    const _Float16* __restrict__ Vt, float* __restrict__ Out) {
  __shared__ __align__(16) _Float16 ldsP[8][16 * 32];
  const int lane = threadIdx.x & 31;
  const int wave = threadIdx.x >> 5;
  const int hl = lane >> 4;
  const int ln = lane & 15;
  const int bh = blockIdx.y;
  const int qbase = blockIdx.x * 128 + wave * 16;

  const _Float16* Qh = Q + (size_t)bh * SLEN * DHEAD;
  const _Float16* Kh = K + (size_t)bh * SLEN * DHEAD;
  const _Float16* Vh = Vt + (size_t)bh * DHEAD * SLEN;

  FragH qf[4];
  {
    const char* qrow = (const char*)(Qh + (size_t)(qbase + ln) * DHEAD) + hl * 16;
#pragma unroll
    for (int dc = 0; dc < 4; ++dc) {
      qf[dc].q[0] = *(const i4*)(qrow + dc * 64);
      qf[dc].q[1] = *(const i4*)(qrow + dc * 64 + 32);
    }
  }

  AccF o[8] = {};
  float m[8], l[8];
#pragma unroll
  for (int i = 0; i < 8; ++i) { m[i] = -1e30f; l[i] = 0.f; }

  for (int kb = 0; kb < qbase + 16; kb += 32) {
    AccF s0 = {}, s1 = {};
#pragma unroll
    for (int dc = 0; dc < 4; ++dc) {
      FragH k0, k1;
      const char* k0p =
          (const char*)(Kh + (size_t)(kb + ln) * DHEAD) + hl * 16 + dc * 64;
      k0.q[0] = *(const i4*)(k0p); k0.q[1] = *(const i4*)(k0p + 32);
      const char* k1p =
          (const char*)(Kh + (size_t)(kb + 16 + ln) * DHEAD) + hl * 16 + dc * 64;
      k1.q[0] = *(const i4*)(k1p); k1.q[1] = *(const i4*)(k1p + 32);
      s0.v = __builtin_amdgcn_wmma_f32_16x16x32_f16(
          false, qf[dc].h, false, k0.h, (short)0, s0.v, false, false);
      s1.v = __builtin_amdgcn_wmma_f32_16x16x32_f16(
          false, qf[dc].h, false, k1.h, (short)0, s1.v, false, false);
    }

    float sc[8];
#pragma unroll
    for (int r = 0; r < 8; ++r) {
      const int qi = qbase + r + hl * 8;
      if (kb + ln > qi)      s0.f[r] = -1e30f;
      if (kb + 16 + ln > qi) s1.f[r] = -1e30f;
      float t = fmaxf(s0.f[r], s1.f[r]);
      t = fmaxf(t, __shfl_xor(t, 1));
      t = fmaxf(t, __shfl_xor(t, 2));
      t = fmaxf(t, __shfl_xor(t, 4));
      t = fmaxf(t, __shfl_xor(t, 8));
      const float nm = fmaxf(m[r], t);
      sc[r] = __expf(m[r] - nm);
      m[r] = nm;
      const float p0 = __expf(s0.f[r] - nm);
      const float p1 = __expf(s1.f[r] - nm);
      s0.f[r] = p0; s1.f[r] = p1;
      float rs = p0 + p1;
      rs += __shfl_xor(rs, 1);
      rs += __shfl_xor(rs, 2);
      rs += __shfl_xor(rs, 4);
      rs += __shfl_xor(rs, 8);
      l[r] = l[r] * sc[r] + rs;
    }
#pragma unroll
    for (int dt = 0; dt < 8; ++dt)
#pragma unroll
      for (int r = 0; r < 8; ++r) o[dt].f[r] *= sc[r];

    _Float16* P = ldsP[wave];
#pragma unroll
    for (int r = 0; r < 8; ++r) {
      const int row = r + hl * 8;
      P[row * 32 + ln]      = (_Float16)s0.f[r];
      P[row * 32 + 16 + ln] = (_Float16)s1.f[r];
    }
    wait_ds0();
    FragH pf;
    {
      const char* pp = (const char*)(P + ln * 32) + hl * 16;
      pf.q[0] = *(const i4*)(pp);
      pf.q[1] = *(const i4*)(pp + 32);
    }

#pragma unroll
    for (int dt = 0; dt < 8; ++dt) {
      FragH vf;
      const char* vp =
          (const char*)(Vh + (size_t)(dt * 16 + ln) * SLEN + kb) + hl * 16;
      vf.q[0] = *(const i4*)(vp); vf.q[1] = *(const i4*)(vp + 32);
      o[dt].v = __builtin_amdgcn_wmma_f32_16x16x32_f16(
          false, pf.h, false, vf.h, (short)0, o[dt].v, false, false);
    }
  }

  const int b = bh >> 4, h = bh & 15;
#pragma unroll
  for (int r = 0; r < 8; ++r) {
    const int row = qbase + r + hl * 8;
    const float inv = 1.f / l[r];
    const size_t base = ((size_t)(b * SLEN + row)) * EMB + h * DHEAD;
#pragma unroll
    for (int dt = 0; dt < 8; ++dt)
      Out[base + dt * 16 + ln] = o[dt].f[r] * inv;
  }
}

extern "C" void kernel_launch(void* const* d_in, const int* in_sizes, int n_in,
                              void* d_out, int out_size, void* d_ws, size_t ws_size,
                              hipStream_t stream) {
  (void)in_sizes; (void)n_in; (void)out_size; (void)ws_size;
  const float* x   = (const float*)d_in[0];
  const float* w_q = (const float*)d_in[1];
  const float* w_k = (const float*)d_in[2];
  const float* w_v = (const float*)d_in[3];
  const float* w_o = (const float*)d_in[4];

  char* ws = (char*)d_ws;
  size_t off = 0;
  auto alloc = [&](size_t bytes) -> void* {
    void* p = ws + off;
    off += (bytes + 255) & ~(size_t)255;
    return p;
  };

  float*       partials = (float*)alloc(4 * 256 * sizeof(float));
  float*       wsum     = (float*)alloc(4 * sizeof(float));
  signed char* wq4      = (signed char*)alloc((size_t)4 * EMB * EMB);
  signed char* xq       = (signed char*)alloc((size_t)TOKENS * EMB);
  float*       ax       = (float*)alloc((size_t)TOKENS * sizeof(float));
  _Float16*    Qf       = (_Float16*)alloc((size_t)TOKENS * EMB * 2);
  _Float16*    Kf       = (_Float16*)alloc((size_t)TOKENS * EMB * 2);
  _Float16*    Vt       = (_Float16*)alloc((size_t)TOKENS * EMB * 2);
  float*       attn     = (float*)alloc((size_t)TOKENS * EMB * sizeof(float));

  wabs_kernel<<<dim3(256, 4), 256, 0, stream>>>(w_q, w_k, w_v, w_o, partials);
  wquant_kernel<<<dim3(1024, 4), 256, 0, stream>>>(w_q, w_k, w_v, w_o,
                                                   partials, wsum, wq4);
  act_quant_kernel<<<TOKENS, 256, 0, stream>>>(x, xq, ax);

  const size_t WB = (size_t)EMB * EMB;
  gemm_i8_kernel<<<dim3(16, 32), 256, 0, stream>>>(xq, ax, wq4 + 0 * WB, wsum,
                                                   0, 0, Qf);
  gemm_i8_kernel<<<dim3(16, 32), 256, 0, stream>>>(xq, ax, wq4 + 1 * WB, wsum,
                                                   1, 1, Kf);
  gemm_i8_kernel<<<dim3(16, 32), 256, 0, stream>>>(xq, ax, wq4 + 2 * WB, wsum,
                                                   2, 2, Vt);
  attn_kernel<<<dim3(16, 32), 256, 0, stream>>>(Qf, Kf, Vt, attn);
  act_quant_kernel<<<TOKENS, 256, 0, stream>>>(attn, xq, ax);
  gemm_i8_kernel<<<dim3(16, 32), 256, 0, stream>>>(xq, ax, wq4 + 3 * WB, wsum,
                                                   3, 3, (float*)d_out);
}
